// MambaEncoder_41300405518819
// MI455X (gfx1250) — matrix operand order repaired and checked by reading, not compiled, not run on hardware
//
#include <hip/hip_runtime.h>
#include <math.h>

// ---------------- problem constants ----------------
constexpr int Bc  = 8;
constexpr int Lc  = 2048;
constexpr int INc = 32;
constexpr int Dc  = 1024;
constexpr int OUTc= 128;
constexpr int Ec  = 2048;      // 2*D
constexpr int NSc = 16;        // state dim
constexpr int Rc  = 64;        // D/16
constexpr int NLc = 2;
constexpr long long Mc = (long long)Bc * Lc;   // 16384 rows
constexpr int XDBL = Rc + 2 * NSc;             // 96

typedef unsigned short u16t;
typedef __attribute__((ext_vector_type(8)))  __bf16 v8bf;
typedef __attribute__((ext_vector_type(16))) __bf16 v16bf;
typedef __attribute__((ext_vector_type(8)))  float  v8f;

// float -> bf16 (round to nearest even)
__device__ __forceinline__ u16t f2bf(float f) {
  union { float f; unsigned u; } v; v.f = f;
  unsigned r = v.u + 0x7FFFu + ((v.u >> 16) & 1u);
  return (u16t)(r >> 16);
}
__device__ __forceinline__ float silu_f(float x) {
  return x / (1.f + __expf(-x));
}

// load a 16-element bf16 fragment as two contiguous 16B chunks (base, base+d)
__device__ __forceinline__ v16bf ldfrag(const u16t* p, int d) {
  v8bf x0 = *reinterpret_cast<const v8bf*>(p);
  v8bf x1 = *reinterpret_cast<const v8bf*>(p + d);
  v16bf r;
  #pragma unroll
  for (int i = 0; i < 8; ++i) { r[i] = x0[i]; r[i + 8] = x1[i]; }
  return r;
}

__device__ __forceinline__ void gemm_epilogue(const v8f& cc, float* C, int ldc,
                                              const float* bias, int row0,
                                              int col, int mode) {
  #pragma unroll
  for (int j = 0; j < 8; ++j) {
    const size_t idx = (size_t)(row0 + j) * ldc + col;
    const float v = cc[j];
    if (mode == 0)      C[idx] = v;
    else if (mode == 1) C[idx] += v;
    else {
      const float xv = v + bias[col];
      C[idx] = (xv > 20.f) ? xv : log1pf(__expf(xv));
    }
  }
}

// ---------------- WMMA GEMM, 32x32 wave tile (for N % 64 != 0) ----------------
// C[M,N] (fp32) = A[M,K](bf16 row-major, stride lda) * B^T (bf16 [N,K], stride ldb)
// mode 0: C = AB ; mode 1: C += AB ; mode 2: C = softplus(AB + bias[n])
__global__ void __launch_bounds__(256)
wmma_gemm_bf16_w32(const u16t* __restrict__ A, int lda,
                   const u16t* __restrict__ Bt, int ldb,
                   float* __restrict__ C, int ldc,
                   const float* __restrict__ bias,
                   int M_, int N_, int K_, int mode)
{
  const int wave   = blockIdx.x * 8 + (threadIdx.x >> 5);
  const int tilesN = N_ >> 5;
  const int tm = wave / tilesN;
  const int tn = wave % tilesN;
  if (tm * 32 >= M_) return;                 // wave-uniform
  const int lane = threadIdx.x & 31;
  const int r    = lane & 15;
  const int hi   = lane >> 4;
  const int m0 = tm * 32, n0 = tn * 32;

  const u16t* a0 = A  + (size_t)(m0 + r)      * lda + hi * 16;
  const u16t* a1 = A  + (size_t)(m0 + 16 + r) * lda + hi * 16;
  const u16t* b0 = Bt + (size_t)(n0 + r)      * ldb + hi * 16;
  const u16t* b1 = Bt + (size_t)(n0 + 16 + r) * ldb + hi * 16;

  v8f c00 = {}, c01 = {}, c10 = {}, c11 = {};

  for (int k = 0; k < K_; k += 32) {
    __builtin_prefetch(a0 + k + 128, 0, 1);
    __builtin_prefetch(b0 + k + 128, 0, 1);
    v16bf fa0 = ldfrag(a0 + k, 8);
    v16bf fa1 = ldfrag(a1 + k, 8);
    v16bf fb0 = ldfrag(b0 + k, 8);
    v16bf fb1 = ldfrag(b1 + k, 8);
    c00 = __builtin_amdgcn_wmma_f32_16x16x32_bf16(false, fa0, false, fb0, (short)0, c00, false, false);
    c01 = __builtin_amdgcn_wmma_f32_16x16x32_bf16(false, fa0, false, fb1, (short)0, c01, false, false);
    c10 = __builtin_amdgcn_wmma_f32_16x16x32_bf16(false, fa1, false, fb0, (short)0, c10, false, false);
    c11 = __builtin_amdgcn_wmma_f32_16x16x32_bf16(false, fa1, false, fb1, (short)0, c11, false, false);
  }

  gemm_epilogue(c00, C, ldc, bias, m0 +  0 + hi * 8, n0 +  0 + r, mode);
  gemm_epilogue(c01, C, ldc, bias, m0 +  0 + hi * 8, n0 + 16 + r, mode);
  gemm_epilogue(c10, C, ldc, bias, m0 + 16 + hi * 8, n0 +  0 + r, mode);
  gemm_epilogue(c11, C, ldc, bias, m0 + 16 + hi * 8, n0 + 16 + r, mode);
}

// ---------------- WMMA GEMM, 32x64 wave tile (N % 64 == 0) ----------------
// Higher register reuse: 12 x b128 loads feed 8 WMMAs per 32-deep k-step.
__global__ void __launch_bounds__(256)
wmma_gemm_bf16_w64(const u16t* __restrict__ A, int lda,
                   const u16t* __restrict__ Bt, int ldb,
                   float* __restrict__ C, int ldc,
                   const float* __restrict__ bias,
                   int M_, int N_, int K_, int mode)
{
  const int wave   = blockIdx.x * 8 + (threadIdx.x >> 5);
  const int tilesN = N_ >> 6;
  const int tm = wave / tilesN;
  const int tn = wave % tilesN;
  if (tm * 32 >= M_) return;                 // wave-uniform
  const int lane = threadIdx.x & 31;
  const int r    = lane & 15;
  const int hi   = lane >> 4;
  const int m0 = tm * 32, n0 = tn * 64;

  const u16t* a0 = A  + (size_t)(m0 + r)      * lda + hi * 16;
  const u16t* a1 = A  + (size_t)(m0 + 16 + r) * lda + hi * 16;
  const u16t* b0 = Bt + (size_t)(n0 + r)      * ldb + hi * 16;
  const u16t* b1 = Bt + (size_t)(n0 + 16 + r) * ldb + hi * 16;
  const u16t* b2 = Bt + (size_t)(n0 + 32 + r) * ldb + hi * 16;
  const u16t* b3 = Bt + (size_t)(n0 + 48 + r) * ldb + hi * 16;

  v8f c00 = {}, c01 = {}, c02 = {}, c03 = {};
  v8f c10 = {}, c11 = {}, c12 = {}, c13 = {};

  for (int k = 0; k < K_; k += 32) {
    __builtin_prefetch(a0 + k + 128, 0, 1);
    __builtin_prefetch(b0 + k + 128, 0, 1);
    __builtin_prefetch(b2 + k + 128, 0, 1);
    v16bf fa0 = ldfrag(a0 + k, 8);
    v16bf fa1 = ldfrag(a1 + k, 8);
    v16bf fb0 = ldfrag(b0 + k, 8);
    v16bf fb1 = ldfrag(b1 + k, 8);
    v16bf fb2 = ldfrag(b2 + k, 8);
    v16bf fb3 = ldfrag(b3 + k, 8);
    c00 = __builtin_amdgcn_wmma_f32_16x16x32_bf16(false, fa0, false, fb0, (short)0, c00, false, false);
    c01 = __builtin_amdgcn_wmma_f32_16x16x32_bf16(false, fa0, false, fb1, (short)0, c01, false, false);
    c02 = __builtin_amdgcn_wmma_f32_16x16x32_bf16(false, fa0, false, fb2, (short)0, c02, false, false);
    c03 = __builtin_amdgcn_wmma_f32_16x16x32_bf16(false, fa0, false, fb3, (short)0, c03, false, false);
    c10 = __builtin_amdgcn_wmma_f32_16x16x32_bf16(false, fa1, false, fb0, (short)0, c10, false, false);
    c11 = __builtin_amdgcn_wmma_f32_16x16x32_bf16(false, fa1, false, fb1, (short)0, c11, false, false);
    c12 = __builtin_amdgcn_wmma_f32_16x16x32_bf16(false, fa1, false, fb2, (short)0, c12, false, false);
    c13 = __builtin_amdgcn_wmma_f32_16x16x32_bf16(false, fa1, false, fb3, (short)0, c13, false, false);
  }

  gemm_epilogue(c00, C, ldc, bias, m0 +  0 + hi * 8, n0 +  0 + r, mode);
  gemm_epilogue(c01, C, ldc, bias, m0 +  0 + hi * 8, n0 + 16 + r, mode);
  gemm_epilogue(c02, C, ldc, bias, m0 +  0 + hi * 8, n0 + 32 + r, mode);
  gemm_epilogue(c03, C, ldc, bias, m0 +  0 + hi * 8, n0 + 48 + r, mode);
  gemm_epilogue(c10, C, ldc, bias, m0 + 16 + hi * 8, n0 +  0 + r, mode);
  gemm_epilogue(c11, C, ldc, bias, m0 + 16 + hi * 8, n0 + 16 + r, mode);
  gemm_epilogue(c12, C, ldc, bias, m0 + 16 + hi * 8, n0 + 32 + r, mode);
  gemm_epilogue(c13, C, ldc, bias, m0 + 16 + hi * 8, n0 + 48 + r, mode);
}

// ---------------- glue kernels ----------------
__global__ void __launch_bounds__(256)
in_proj_kernel(const float* __restrict__ x, const float* __restrict__ Wi,
               const float* __restrict__ bi, float* __restrict__ h)
{
  const size_t i = (size_t)blockIdx.x * 256 + threadIdx.x;   // < M*D
  const size_t m = i / Dc;
  const int    d = (int)(i % Dc);
  const float* xr = x + m * INc;
  float acc = bi[d];
  #pragma unroll
  for (int k = 0; k < INc; ++k) acc += xr[k] * Wi[(size_t)k * Dc + d];
  h[i] = acc;
}

__global__ void __launch_bounds__(256)
rmsnorm_bf16_kernel(const float* __restrict__ h, const float* __restrict__ w,
                    u16t* __restrict__ out)
{
  __shared__ float red[256];
  const size_t row = blockIdx.x;
  const float* hr = h + row * Dc;
  float ss = 0.f;
  for (int d = threadIdx.x; d < Dc; d += 256) { float v = hr[d]; ss += v * v; }
  red[threadIdx.x] = ss; __syncthreads();
  for (int s = 128; s > 0; s >>= 1) {
    if (threadIdx.x < s) red[threadIdx.x] += red[threadIdx.x + s];
    __syncthreads();
  }
  const float rs = rsqrtf(red[0] / (float)Dc + 1e-5f);
  for (int d = threadIdx.x; d < Dc; d += 256)
    out[row * Dc + d] = f2bf(hr[d] * rs * w[d]);
}

// transpose-convert weight [K,N] fp32 -> [N,K] bf16
__global__ void __launch_bounds__(256)
wT_bf16_kernel(const float* __restrict__ W, u16t* __restrict__ Wt, int Kd, int Nd)
{
  const size_t i = (size_t)blockIdx.x * 256 + threadIdx.x;   // < Kd*Nd
  const int k = (int)(i / Nd);
  const int n = (int)(i % Nd);
  Wt[(size_t)n * Kd + k] = f2bf(W[i]);
}

// causal depthwise conv (K=4) + bias + SiLU ; xin = xz[..., :E]
__global__ void __launch_bounds__(256)
conv_silu_kernel(const float* __restrict__ xz, const float* __restrict__ cw,
                 const float* __restrict__ cb, float* __restrict__ xc,
                 u16t* __restrict__ xc16)
{
  const size_t i = (size_t)blockIdx.x * 256 + threadIdx.x;   // < M*E
  const int    e = (int)(i % Ec);
  const size_t m = i / Ec;
  const int    t = (int)(m % Lc);
  float acc = cb[e];
  #pragma unroll
  for (int j = 0; j < 4; ++j) {
    const int tp = t - 3 + j;
    if (tp >= 0) acc += cw[e * 4 + j] * xz[(m - 3 + j) * (size_t)(2 * Ec) + e];
  }
  const float v = silu_f(acc);
  xc[i]   = v;
  xc16[i] = f2bf(v);
}

__global__ void __launch_bounds__(256)
f2bf_kernel(const float* __restrict__ in, u16t* __restrict__ out)
{
  const size_t i = (size_t)blockIdx.x * 256 + threadIdx.x;
  out[i] = f2bf(in[i]);
}

// selective scan: one thread per (b,e); dt buffer overwritten with y in place.
// Bm/Cm rows (128B/step) are double-buffered through LDS via async-to-LDS
// loads issued one timestep ahead (wave 0, lanes 0..7), retired with
// s_wait_asynccnt before the block barrier.
__global__ void __launch_bounds__(256)
scan_kernel(float* __restrict__ dty, const float* __restrict__ xc,
            const float* __restrict__ xdbl, const float* __restrict__ A_log,
            const float* __restrict__ Dp)
{
  const int b = blockIdx.x;
  const int e = blockIdx.y * 256 + threadIdx.x;
  __shared__ __align__(16) float sBC[2][32];

  float An[NSc];
  #pragma unroll
  for (int n = 0; n < NSc; ++n) An[n] = -__expf(A_log[e * NSc + n]);
  float hs[NSc];
  #pragma unroll
  for (int n = 0; n < NSc; ++n) hs[n] = 0.f;
  const float dpe = Dp[e];

  const float* bc_base = xdbl + (size_t)b * Lc * XDBL + Rc;

  // prime buffer 0 with t=0 (8 lanes x b128 = 128 bytes)
  if (threadIdx.x < 8) {
    const float* src = bc_base + (size_t)threadIdx.x * 4;
    unsigned dst = (unsigned)(size_t)&sBC[0][threadIdx.x * 4];
    asm volatile("global_load_async_to_lds_b128 %0, %1, off"
                 :: "v"(dst), "v"(src) : "memory");
  }

  for (int t = 0; t < Lc; ++t) {
    asm volatile("s_wait_asynccnt 0x0" ::: "memory");
    __syncthreads();
    if ((t + 1 < Lc) && threadIdx.x < 8) {
      const float* src = bc_base + (size_t)(t + 1) * XDBL + threadIdx.x * 4;
      unsigned dst = (unsigned)(size_t)&sBC[(t + 1) & 1][threadIdx.x * 4];
      asm volatile("global_load_async_to_lds_b128 %0, %1, off"
                   :: "v"(dst), "v"(src) : "memory");
    }
    const float* bc = sBC[t & 1];
    const size_t idx = ((size_t)b * Lc + t) * Ec + e;
    const float dtv = dty[idx];
    const float xv  = xc[idx];
    const float dx  = dtv * xv;
    float y = 0.f;
    #pragma unroll
    for (int n = 0; n < NSc; ++n) {
      const float dA = __expf(dtv * An[n]);
      hs[n] = hs[n] * dA + dx * bc[n];
      y += hs[n] * bc[NSc + n];
    }
    dty[idx] = y + xv * dpe;   // overwrite dt with y (same thread, same address)
  }
}

// g16 = bf16( y * silu(z) ), z = xz[..., E:]
__global__ void __launch_bounds__(256)
gate_kernel(const float* __restrict__ y, const float* __restrict__ xz,
            u16t* __restrict__ g16)
{
  const size_t i = (size_t)blockIdx.x * 256 + threadIdx.x;   // < M*E
  const size_t m = i / Ec;
  const int    e = (int)(i % Ec);
  const float z = xz[m * (size_t)(2 * Ec) + Ec + e];
  g16[i] = f2bf(y[i] * silu_f(z));
}

__global__ void __launch_bounds__(256)
zero_kernel(float* __restrict__ p)
{
  p[(size_t)blockIdx.x * 256 + threadIdx.x] = 0.f;
}

// final rmsnorm + mean over L accumulated into acc[b, d]
__global__ void __launch_bounds__(256)
rms_mean_kernel(const float* __restrict__ h, const float* __restrict__ w,
                float* __restrict__ acc)
{
  __shared__ float red[256];
  const size_t row = blockIdx.x;
  const int b = (int)(row / Lc);
  const float* hr = h + row * Dc;
  float ss = 0.f;
  for (int d = threadIdx.x; d < Dc; d += 256) { float v = hr[d]; ss += v * v; }
  red[threadIdx.x] = ss; __syncthreads();
  for (int s = 128; s > 0; s >>= 1) {
    if (threadIdx.x < s) red[threadIdx.x] += red[threadIdx.x + s];
    __syncthreads();
  }
  const float rs = rsqrtf(red[0] / (float)Dc + 1e-5f) * (1.f / (float)Lc);
  for (int d = threadIdx.x; d < Dc; d += 256)
    atomicAdd(&acc[(size_t)b * Dc + d], hr[d] * rs * w[d]);
}

// out[b,o] = acc[b,:] @ Wo[:,o] + bo[o]
__global__ void __launch_bounds__(256)
out_proj_kernel(const float* __restrict__ acc, const float* __restrict__ Wo,
                const float* __restrict__ bo, float* __restrict__ out)
{
  const int i = blockIdx.x * 256 + threadIdx.x;   // < B*OUT
  const int b = i / OUTc;
  const int o = i % OUTc;
  float s = bo[o];
  for (int d = 0; d < Dc; ++d) s += acc[(size_t)b * Dc + d] * Wo[(size_t)d * OUTc + o];
  out[i] = s;
}

// ---------------- launch ----------------
extern "C" void kernel_launch(void* const* d_in, const int* in_sizes, int n_in,
                              void* d_out, int out_size, void* d_ws, size_t ws_size,
                              hipStream_t stream)
{
  const float* x      = (const float*)d_in[0];
  const float* Wi     = (const float*)d_in[1];
  const float* bi     = (const float*)d_in[2];
  const float* Win    = (const float*)d_in[3];
  const float* conv_w = (const float*)d_in[4];
  const float* conv_b = (const float*)d_in[5];
  const float* Wx     = (const float*)d_in[6];
  const float* Wdt    = (const float*)d_in[7];
  const float* bdt    = (const float*)d_in[8];
  const float* A_log  = (const float*)d_in[9];
  const float* Dp     = (const float*)d_in[10];
  const float* Wout   = (const float*)d_in[11];
  const float* norm_w = (const float*)d_in[12];
  const float* fnorm_w= (const float*)d_in[13];
  const float* Wo     = (const float*)d_in[14];
  const float* bo     = (const float*)d_in[15];
  float* out = (float*)d_out;

  // workspace carve-up
  char* ws = (char*)d_ws;
  size_t off = 0;
  auto carve = [&](size_t bytes) -> void* {
    void* p = ws + off;
    off += (bytes + 255) & ~(size_t)255;
    return p;
  };
  float* h      = (float*)carve(Mc * Dc * 4);
  u16t*  u16    = (u16t*) carve(Mc * Dc * 2);
  float* xz     = (float*)carve(Mc * 2 * Ec * 4);
  float* xc     = (float*)carve(Mc * Ec * 4);
  u16t*  xc16   = (u16t*) carve(Mc * Ec * 2);
  float* xdbl   = (float*)carve(Mc * XDBL * 4);
  u16t*  xdbl16 = (u16t*) carve(Mc * XDBL * 2);
  float* dty    = (float*)carve(Mc * Ec * 4);
  u16t*  g16    = (u16t*) carve(Mc * Ec * 2);
  u16t*  wt     = (u16t*) carve((size_t)Dc * 2 * Ec * 2);   // largest weight (Win)
  float* acc    = (float*)carve((size_t)Bc * Dc * 4);

  const int MD_blocks = (int)((Mc * Dc) / 256);
  const int ME_blocks = (int)((Mc * Ec) / 256);
  const int tilesM    = (int)(Mc / 32);   // 512

  // h = x @ Wi + bi
  in_proj_kernel<<<MD_blocks, 256, 0, stream>>>(x, Wi, bi, h);

  for (int l = 0; l < NLc; ++l) {
    const float* Win_l  = Win   + (size_t)l * Dc * 2 * Ec;
    const float* cw_l   = conv_w+ (size_t)l * Ec * 4;
    const float* cb_l   = conv_b+ (size_t)l * Ec;
    const float* Wx_l   = Wx    + (size_t)l * Ec * XDBL;
    const float* Wdt_l  = Wdt   + (size_t)l * Rc * Ec;
    const float* bdt_l  = bdt   + (size_t)l * Ec;
    const float* Alog_l = A_log + (size_t)l * Ec * NSc;
    const float* Dp_l   = Dp    + (size_t)l * Ec;
    const float* Wout_l = Wout  + (size_t)l * Ec * Dc;
    const float* nw_l   = norm_w+ (size_t)l * Dc;

    // u = bf16(rmsnorm(h))
    rmsnorm_bf16_kernel<<<(int)Mc, 256, 0, stream>>>(h, nw_l, u16);

    // xz = u @ Win  (M=16384, N=4096, K=1024)
    wT_bf16_kernel<<<(int)(((size_t)Dc * 2 * Ec) / 256), 256, 0, stream>>>(Win_l, wt, Dc, 2 * Ec);
    wmma_gemm_bf16_w64<<<(tilesM * (2 * Ec / 64)) / 8, 256, 0, stream>>>(
        u16, Dc, wt, Dc, xz, 2 * Ec, nullptr, (int)Mc, 2 * Ec, Dc, 0);

    // xc = silu(dwconv(xz[:, :E]) + cb)
    conv_silu_kernel<<<ME_blocks, 256, 0, stream>>>(xz, cw_l, cb_l, xc, xc16);

    // x_dbl = xc @ Wx  (N=96, K=2048)
    wT_bf16_kernel<<<(int)(((size_t)Ec * XDBL) / 256), 256, 0, stream>>>(Wx_l, wt, Ec, XDBL);
    wmma_gemm_bf16_w32<<<(tilesM * (XDBL / 32)) / 8, 256, 0, stream>>>(
        xc16, Ec, wt, Ec, xdbl, XDBL, nullptr, (int)Mc, XDBL, Ec, 0);

    // dt = softplus(x_dbl[:, :R] @ Wdt + bdt)  (N=2048, K=64)
    f2bf_kernel<<<(int)((Mc * XDBL) / 256), 256, 0, stream>>>(xdbl, xdbl16);
    wT_bf16_kernel<<<(int)(((size_t)Rc * Ec) / 256), 256, 0, stream>>>(Wdt_l, wt, Rc, Ec);
    wmma_gemm_bf16_w64<<<(tilesM * (Ec / 64)) / 8, 256, 0, stream>>>(
        xdbl16, XDBL, wt, Rc, dty, Ec, bdt_l, (int)Mc, Ec, Rc, 2);

    // selective scan: y written over dt in place
    scan_kernel<<<dim3(Bc, Ec / 256), 256, 0, stream>>>(dty, xc, xdbl, Alog_l, Dp_l);

    // g = bf16(y * silu(z))
    gate_kernel<<<ME_blocks, 256, 0, stream>>>(dty, xz, g16);

    // h += g @ Wout  (N=1024, K=2048, residual add)
    wT_bf16_kernel<<<(int)(((size_t)Ec * Dc) / 256), 256, 0, stream>>>(Wout_l, wt, Ec, Dc);
    wmma_gemm_bf16_w64<<<(tilesM * (Dc / 64)) / 8, 256, 0, stream>>>(
        g16, Ec, wt, Ec, h, Dc, nullptr, (int)Mc, Dc, Ec, 1);
  }

  // final rmsnorm, mean over L, output projection
  zero_kernel<<<(Bc * Dc) / 256, 256, 0, stream>>>(acc);
  rms_mean_kernel<<<(int)Mc, 256, 0, stream>>>(h, fnorm_w, acc);
  out_proj_kernel<<<(Bc * OUTc) / 256, 256, 0, stream>>>(acc, Wo, bo, out);
}
